// LinearBase_56195352101388
// MI455X (gfx1250) — matrix operand order repaired and checked by reading, not compiled, not run on hardware
//
#include <hip/hip_runtime.h>

typedef __attribute__((ext_vector_type(16))) _Float16 v16h;
typedef __attribute__((ext_vector_type(2)))  _Float16 h2;
typedef __attribute__((ext_vector_type(8)))  float    v8f;

#define I_DIM 4096
#define O_DIM 11008
#define O8    (O_DIM / 8)      // 1376 packed int32 per input row
#define GS    128              // quant group size
#define BM    128
#define BN    128
#define BK    32               // one f16-WMMA K step
#define STEPS (I_DIM / BK)     // 128 k-slabs
#define LDA   40               // LDS row stride in halves (32 + 8 pad)
#define LDB   40

union Frag16 { uint4 u[2]; v16h v; };
union Pack16 { _Float16 h[16]; uint4 u[2]; };
union PairU  { unsigned u; h2 h; };

__global__ __launch_bounds__(256)
void w4_f16_gemm_wmma(const float* __restrict__ x,
                      const int*   __restrict__ qweight,
                      const float* __restrict__ scales,
                      const int*   __restrict__ qzeros,
                      float*       __restrict__ out)
{
    // double-buffered stages: 2 x (10 KB + 10 KB) of the 320 KB WGP LDS pool
    __shared__ __align__(16) _Float16 As[2][BM * LDA];
    __shared__ __align__(16) _Float16 Bs[2][BN * LDB];   // n-major (B^T)

    const int tid  = threadIdx.x;
    const int lane = tid & 31;
    const int wave = tid >> 5;
    const int wm   = wave >> 1;          // 0..3 : 32-row band
    const int wn   = wave & 1;           // 0..1 : 64-col band

    const int m0   = blockIdx.y * BM;
    const int n0   = blockIdx.x * BN;
    const int n0_8 = n0 >> 3;

    // A-fill: thread -> (row, 16-wide k half)
    const int arow  = tid >> 1;          // 0..127
    const int ahalf = (tid & 1) * 16;    // 0 or 16

    // B-fill: thread -> 8 contiguous k rows of one packed dword; nibble pair
    // (j, j+4) of that dword = output columns c0, c0+4 (u16-lane pairing)
    const int o8l = tid & 15;            // packed int32 within 128-col tile
    const int kc  = (tid >> 4) & 3;      // 8-row k chunk: k = kc*8 .. +7
    const int jp  = tid >> 6;            // 0..3 -> nibbles jp and jp+4
    const int sh4 = jp * 4;              // shift aligning nibble pair to bits 0 / 16
    const int c0  = o8l * 8 + jp;        // local output column (lo16 lane)
    const int c1  = c0 + 4;              // local output column (hi16 lane)

    // WMMA 16-bit operand addressing (wave32 layouts, ISA 7.12.2)
    const int hi   = lane >> 4;
    const int mloc = lane & 15;
    const int akb  = hi * 8;             // A: 8 halves @akb, 8 @akb+16
    const int bkb  = hi * 8;             // B: 8 halves @bkb, 8 @bkb+16

    h2 sc2 = {(_Float16)0.f, (_Float16)0.f};   // per-column scale (f16x2)
    h2 bz2 = {(_Float16)0.f, (_Float16)0.f};   // per-column -s*(1024+z) (f16x2)

    auto stage = [&](int s, int buf) {
        const int k0 = s * BK;

        // group-boundary reload of dequant constants (uniform branch)
        if ((s & 3) == 0) {
            const int g = s >> 2;
            const unsigned qz = (unsigned)qzeros[g * O8 + n0_8 + o8l];
            const float z0 = (float)((qz >> (sh4     )) & 0xFu);
            const float z1 = (float)((qz >> (sh4 + 16)) & 0xFu);
            const float s0 = scales[(size_t)g * O_DIM + n0 + c0];
            const float s1 = scales[(size_t)g * O_DIM + n0 + c1];
            sc2[0] = (_Float16)s0;
            sc2[1] = (_Float16)s1;
            bz2[0] = (_Float16)(-s0 * (1024.f + z0));
            bz2[1] = (_Float16)(-s1 * (1024.f + z1));
        }

        // ---- stage A: x[m0+arow][k0+ahalf .. +15] fp32 -> f16 -> LDS
        {
            const float* xp = x + (size_t)(m0 + arow) * I_DIM + k0 + ahalf;
            const float4* xv = (const float4*)xp;
            float4 f0 = xv[0], f1 = xv[1], f2 = xv[2], f3 = xv[3];
            __builtin_prefetch(xp + BK, 0, 1);      // k-slab s+2 -> global_prefetch_b8
            Pack16 p;
            p.h[ 0]=(_Float16)f0.x; p.h[ 1]=(_Float16)f0.y; p.h[ 2]=(_Float16)f0.z; p.h[ 3]=(_Float16)f0.w;
            p.h[ 4]=(_Float16)f1.x; p.h[ 5]=(_Float16)f1.y; p.h[ 6]=(_Float16)f1.z; p.h[ 7]=(_Float16)f1.w;
            p.h[ 8]=(_Float16)f2.x; p.h[ 9]=(_Float16)f2.y; p.h[10]=(_Float16)f2.z; p.h[11]=(_Float16)f2.w;
            p.h[12]=(_Float16)f3.x; p.h[13]=(_Float16)f3.y; p.h[14]=(_Float16)f3.z; p.h[15]=(_Float16)f3.w;
            uint4* dst = (uint4*)(&As[buf][arow * LDA + ahalf]);
            dst[0] = p.u[0];
            dst[1] = p.u[1];
        }

        // ---- stage B: mantissa-injection dequant, one v_pk_fma_f16 per 2 values
        {
            unsigned qw[8];
            const int* qp = qweight + (size_t)(k0 + kc * 8) * O8 + n0_8 + o8l;
#pragma unroll
            for (int r = 0; r < 8; ++r) qw[r] = (unsigned)qp[(size_t)r * O8];

            unsigned d[8];
#pragma unroll
            for (int r = 0; r < 8; ++r) {
                PairU t;
                t.u = ((qw[r] >> sh4) & 0x000F000Fu) | 0x64006400u;  // f16x2: 1024+nib
                PairU w;
                w.h = t.h * sc2 + bz2;                               // v_pk_fma_f16
                d[r] = w.u;
            }
            // transpose (col, col+4) pairs -> per-column k-contiguous packs
            unsigned p0[4], p1[4];
#pragma unroll
            for (int q = 0; q < 4; ++q) {
                p0[q] = __builtin_amdgcn_perm(d[2*q+1], d[2*q], 0x05040100u); // lo16 lanes
                p1[q] = __builtin_amdgcn_perm(d[2*q+1], d[2*q], 0x07060302u); // hi16 lanes
            }
            *(uint4*)(&Bs[buf][c0 * LDB + kc * 8]) = make_uint4(p0[0], p0[1], p0[2], p0[3]);
            *(uint4*)(&Bs[buf][c1 * LDB + kc * 8]) = make_uint4(p1[0], p1[1], p1[2], p1[3]);
        }
    };

    const v8f vzero = {0.f,0.f,0.f,0.f,0.f,0.f,0.f,0.f};
    v8f acc[2][4];
#pragma unroll
    for (int i = 0; i < 2; ++i)
#pragma unroll
        for (int j = 0; j < 4; ++j) acc[i][j] = vzero;

    stage(0, 0);

    for (int s = 0; s < STEPS; ++s) {
        const int buf = s & 1;
        __syncthreads();   // stage(s) writes visible; reads of buf^1 finished

        // fragment ds_reads FIRST so the pre-WMMA s_wait_dscnt excludes the
        // stage stores issued below (DS completes in order)
        Frag16 a[2], b[4];
#pragma unroll
        for (int fm = 0; fm < 2; ++fm) {
            const _Float16* ap = &As[buf][(wm * 32 + fm * 16 + mloc) * LDA];
            a[fm].u[0] = *(const uint4*)(ap + akb);
            a[fm].u[1] = *(const uint4*)(ap + akb + 16);
        }
#pragma unroll
        for (int fn = 0; fn < 4; ++fn) {
            const _Float16* bp = &Bs[buf][(wn * 64 + fn * 16 + mloc) * LDB];
            b[fn].u[0] = *(const uint4*)(bp + bkb);
            b[fn].u[1] = *(const uint4*)(bp + bkb + 16);
        }

        // overlap: global loads + pk-fma dequant + LDS stores of slab s+1
        // co-execute with this slab's WMMAs (XDL tracked as TRANS)
        if (s + 1 < STEPS) stage(s + 1, buf ^ 1);

#pragma unroll
        for (int fm = 0; fm < 2; ++fm)
#pragma unroll
            for (int fn = 0; fn < 4; ++fn)
                acc[fm][fn] = __builtin_amdgcn_wmma_f32_16x16x32_f16(
                    false, a[fm].v, false, b[fn].v,
                    (short)0, acc[fm][fn], false, false);
    }

    // ---- epilogue: C/D layout: VGPR i -> row i + hi*8, col = mloc
#pragma unroll
    for (int fm = 0; fm < 2; ++fm) {
        const int rbase = m0 + wm * 32 + fm * 16 + hi * 8;
#pragma unroll
        for (int fn = 0; fn < 4; ++fn) {
            const int col = n0 + wn * 64 + fn * 16 + mloc;
#pragma unroll
            for (int i = 0; i < 8; ++i)
                out[(size_t)(rbase + i) * O_DIM + col] = acc[fm][fn][i];
        }
    }
}

extern "C" void kernel_launch(void* const* d_in, const int* in_sizes, int n_in,
                              void* d_out, int out_size, void* d_ws, size_t ws_size,
                              hipStream_t stream) {
    (void)n_in; (void)d_ws; (void)ws_size; (void)out_size;
    const float* x       = (const float*)d_in[0];
    const int*   qweight = (const int*)d_in[1];
    const float* scales  = (const float*)d_in[2];
    const int*   qzeros  = (const int*)d_in[3];
    float*       out     = (float*)d_out;

    const int n_rows = in_sizes[0] / I_DIM;          // 4096
    dim3 grid(O_DIM / BN, n_rows / BM);              // 86 x 32
    dim3 block(256);                                 // 8 wave32
    w4_f16_gemm_wmma<<<grid, block, 0, stream>>>(x, qweight, scales, qzeros, out);
}
